// MultiHeadsCrossLinearAttention_23321672417458
// MI455X (gfx1250) — matrix operand order repaired and checked by reading, not compiled, not run on hardware
//
#include <hip/hip_runtime.h>
#include <hip/hip_bf16.h>

// ---------------------------------------------------------------------------
// Types & WMMA helpers (CDNA5 / gfx1250, wave32)
// ---------------------------------------------------------------------------
typedef __attribute__((ext_vector_type(16))) __bf16 bf16x16;
typedef __attribute__((ext_vector_type(8)))  __bf16 bf16x8;
typedef __attribute__((ext_vector_type(8)))  float  f32x8;
typedef __attribute__((ext_vector_type(4)))  float  f32x4;

union BF16Frag { bf16x16 v; bf16x8 h[2]; };

__device__ __forceinline__ f32x8 wmma_bf16(bf16x16 a, bf16x16 b, f32x8 c) {
  // D = A(16x32 bf16) x B(32x16 bf16) + C(16x16 f32)
  return __builtin_amdgcn_wmma_f32_16x16x32_bf16(false, a, false, b, (short)0, c,
                                                 false, false);
}

// A fragment, 16x32 bf16, row-major source, ld = row stride (elements).
// ISA layout: lanes 0-15 hold row M=lane, K={0..7} in v0-3 and K={16..23} in v4-7;
// lanes 16-31 hold row M=lane-16 with K offset by +8.
__device__ __forceinline__ bf16x16 load_a_frag(const __bf16* base, int ld, int lane) {
  const __bf16* p = base + (size_t)(lane & 15) * ld + ((lane >> 4) << 3);
  BF16Frag u;
  u.h[0] = *(const bf16x8*)(p);
  u.h[1] = *(const bf16x8*)(p + 16);
  return u.v;
}

// B fragment, 32x16 bf16, source stored COLUMN-major (i.e. base is [N][K] row-major,
// column n contiguous in K). Lanes 0-15: col N=lane, K={0..7,16..23}; lanes 16-31: K offset by +8.
__device__ __forceinline__ bf16x16 load_b_frag(const __bf16* base, int ld, int lane) {
  const __bf16* p = base + (size_t)(lane & 15) * ld + ((lane >> 4) << 3);
  BF16Frag u;
  u.h[0] = *(const bf16x8*)(p);
  u.h[1] = *(const bf16x8*)(p + 16);
  return u.v;
}

// C/D fragment f32 16x16: lane L, vgpr r -> row M = r + 8*(L/16), col N = L%16.

// ---------------------------------------------------------------------------
// Problem constants
// ---------------------------------------------------------------------------
#define B_    8
#define QC_   256
#define HID_  512
#define N_    4096
#define DH_   64
#define HEADS_ 8

// ---------------------------------------------------------------------------
// K0: fp32 -> bf16 weight conversion
// ---------------------------------------------------------------------------
__global__ void k0_cvt_bf16(const float* __restrict__ s, __bf16* __restrict__ d, int n) {
  int i = blockIdx.x * 256 + threadIdx.x;
  if (i < n) d[i] = (__bf16)s[i];
}

// ---------------------------------------------------------------------------
// K1: rmsnorm over channels; write xn fp32 [b][c][n] (residual) and
//     t bf16 [b][n][c] (WMMA A operand for QKV GEMM).
// grid (N_/256, B_), block 256; one pixel per thread.
// ---------------------------------------------------------------------------
__global__ void k1_rmsnorm_t(const float* __restrict__ x, const float* __restrict__ g_norm,
                             float* __restrict__ xn, __bf16* __restrict__ t) {
  const int b = blockIdx.y;
  const int n = blockIdx.x * 256 + threadIdx.x;
  const float* xb = x + (size_t)b * QC_ * N_;
  float ss = 0.f;
  for (int c = 0; c < QC_; ++c) { float v = xb[(size_t)c * N_ + n]; ss += v * v; }
  const float scale = 16.0f / fmaxf(sqrtf(ss), 1e-12f);   // sqrt(256)=16
  float* xnb = xn + (size_t)b * QC_ * N_;
  __bf16* tb = t + ((size_t)b * N_ + n) * QC_;
  bf16x8 pack;
  for (int c = 0; c < QC_; ++c) {
    float v = xb[(size_t)c * N_ + n] * scale * g_norm[c];
    xnb[(size_t)c * N_ + n] = v;
    pack[c & 7] = (__bf16)v;
    if ((c & 7) == 7) *(bf16x8*)(tb + (c - 7)) = pack;
  }
}

// ---------------------------------------------------------------------------
// K2: Q/K/V = t[4096x256] @ W^T[256x512] per batch, bf16 WMMA.
// grid (32, 8, 24): x = 128-row tile, y = 64-col tile, z = b*3 + {q,k,v}
// block 256 (8 waves); wave computes 16 rows x 64 cols (4 WMMA tiles), K loop 256/32.
// Epilogues: q -> fused softmax over d=64; k -> transposed fp32; v -> transposed bf16.
// ---------------------------------------------------------------------------
__global__ void k2_qkv(const __bf16* __restrict__ t,
                       const __bf16* __restrict__ Wq, const __bf16* __restrict__ Wk,
                       const __bf16* __restrict__ Wv,
                       __bf16* __restrict__ q_s, float* __restrict__ kT,
                       __bf16* __restrict__ vT) {
  const int b = blockIdx.z / 3, m = blockIdx.z % 3;
  const int lane = threadIdx.x & 31, wave = threadIdx.x >> 5;
  const int row0 = blockIdx.x * 128 + wave * 16;
  const int col0 = blockIdx.y * 64;
  const __bf16* W = (m == 0) ? Wq : (m == 1) ? Wk : Wv;
  const __bf16* A0 = t + ((size_t)b * N_ + row0) * QC_;

  f32x8 acc[4] = {};
  for (int k = 0; k < QC_; k += 32) {
    bf16x16 a = load_a_frag(A0 + k, QC_, lane);
#pragma unroll
    for (int j = 0; j < 4; ++j) {
      bf16x16 bb = load_b_frag(W + (size_t)(col0 + j * 16) * QC_ + k, QC_, lane);
      acc[j] = wmma_bf16(a, bb, acc[j]);
    }
  }

  const int rbase = ((lane >> 4) << 3);
  if (m == 0) {
    // softmax over the 64 cols of this tile (== one head's d dimension)
    __bf16* qp = q_s + (size_t)b * N_ * HID_;
#pragma unroll
    for (int r = 0; r < 8; ++r) {
      float v0 = acc[0][r], v1 = acc[1][r], v2 = acc[2][r], v3 = acc[3][r];
      float mx = fmaxf(fmaxf(v0, v1), fmaxf(v2, v3));
      for (int s = 1; s < 16; s <<= 1) mx = fmaxf(mx, __shfl_xor(mx, s, 32));
      float e0 = __expf(v0 - mx), e1 = __expf(v1 - mx);
      float e2 = __expf(v2 - mx), e3 = __expf(v3 - mx);
      float sum = e0 + e1 + e2 + e3;
      for (int s = 1; s < 16; s <<= 1) sum += __shfl_xor(sum, s, 32);
      float inv = 1.0f / sum;
      size_t ro = (size_t)(row0 + rbase + r) * HID_ + col0 + (lane & 15);
      qp[ro]      = (__bf16)(e0 * inv);
      qp[ro + 16] = (__bf16)(e1 * inv);
      qp[ro + 32] = (__bf16)(e2 * inv);
      qp[ro + 48] = (__bf16)(e3 * inv);
    }
  } else if (m == 1) {
    // raw k, transposed fp32 [hid][n] for the column softmax kernel
    float* kp = kT + (size_t)b * HID_ * N_;
    const int n0 = row0 + rbase;
#pragma unroll
    for (int j = 0; j < 4; ++j) {
      int col = col0 + j * 16 + (lane & 15);
      f32x4 lo = {acc[j][0], acc[j][1], acc[j][2], acc[j][3]};
      f32x4 hi = {acc[j][4], acc[j][5], acc[j][6], acc[j][7]};
      *(f32x4*)(kp + (size_t)col * N_ + n0)     = lo;
      *(f32x4*)(kp + (size_t)col * N_ + n0 + 4) = hi;
    }
  } else {
    // v, transposed bf16 [hid][n]: direct WMMA B operand for attn GEMM
    __bf16* vp = vT + (size_t)b * HID_ * N_;
    const int n0 = row0 + rbase;
#pragma unroll
    for (int j = 0; j < 4; ++j) {
      int col = col0 + j * 16 + (lane & 15);
      bf16x8 p;
#pragma unroll
      for (int r = 0; r < 8; ++r) p[r] = (__bf16)acc[j][r];
      *(bf16x8*)(vp + (size_t)col * N_ + n0) = p;
    }
  }
}

// ---------------------------------------------------------------------------
// K3: softmax of k over n (per column), from kT fp32 rows (contiguous).
// Writes Et bf16 [b][hid][n] = exp(k - max)/sum * DH^-0.5 (q scale folded in).
// grid 4096 (= b*512 + col), block 256.
// ---------------------------------------------------------------------------
__global__ void k3_ksoftmax(const float* __restrict__ kT, __bf16* __restrict__ Et) {
  const int bc = blockIdx.x;                 // b*512 + col
  const float* src = kT + (size_t)bc * N_;
  __bf16* dst = Et + (size_t)bc * N_;
  __shared__ float red[256];
  const int tid = threadIdx.x;
  float vals[16];
  float mx = -INFINITY;
#pragma unroll
  for (int i = 0; i < 16; ++i) { vals[i] = src[tid + 256 * i]; mx = fmaxf(mx, vals[i]); }
  red[tid] = mx; __syncthreads();
  for (int s = 128; s > 0; s >>= 1) { if (tid < s) red[tid] = fmaxf(red[tid], red[tid + s]); __syncthreads(); }
  mx = red[0]; __syncthreads();
  float sum = 0.f;
#pragma unroll
  for (int i = 0; i < 16; ++i) { vals[i] = __expf(vals[i] - mx); sum += vals[i]; }
  red[tid] = sum; __syncthreads();
  for (int s = 128; s > 0; s >>= 1) { if (tid < s) red[tid] += red[tid + s]; __syncthreads(); }
  const float inv = 0.125f / red[0];         // * DH^-0.5
#pragma unroll
  for (int i = 0; i < 16; ++i) dst[tid + 256 * i] = (__bf16)(vals[i] * inv);
}

// ---------------------------------------------------------------------------
// K4: attn[d][e] = sum_n Et[n,d] * V[n,e] per (b,h). A = Et rows, B = vT rows.
// grid 64 (b*8+h), block 256; wave handles 1 d-tile x 2 e-tiles, full K=4096.
// Stores attnT bf16 [b][h][e][d] (WMMA B operand for K5).
// ---------------------------------------------------------------------------
__global__ void k4_attn(const __bf16* __restrict__ Et, const __bf16* __restrict__ vT,
                        __bf16* __restrict__ attnT) {
  const int b = blockIdx.x >> 3, h = blockIdx.x & 7;
  const int lane = threadIdx.x & 31, wave = threadIdx.x >> 5;
  const int dtile = (wave & 3) * 16;
  const int etile = (wave >> 2) * 32;
  const __bf16* Eb = Et + (size_t)(b * HID_ + h * DH_ + dtile) * N_;
  const __bf16* Vb = vT + (size_t)(b * HID_ + h * DH_ + etile) * N_;
  f32x8 acc0 = {}, acc1 = {};
  for (int k = 0; k < N_; k += 32) {
    bf16x16 a  = load_a_frag(Eb + k, N_, lane);
    bf16x16 b0 = load_b_frag(Vb + k, N_, lane);
    bf16x16 b1 = load_b_frag(Vb + (size_t)16 * N_ + k, N_, lane);
    acc0 = wmma_bf16(a, b0, acc0);
    acc1 = wmma_bf16(a, b1, acc1);
  }
  __bf16* At = attnT + (size_t)(b * HEADS_ + h) * DH_ * DH_;
  const int d0 = dtile + ((lane >> 4) << 3);
  {
    int e = etile + (lane & 15);
    bf16x8 p;
#pragma unroll
    for (int r = 0; r < 8; ++r) p[r] = (__bf16)acc0[r];
    *(bf16x8*)(At + (size_t)e * DH_ + d0) = p;
    e = etile + 16 + (lane & 15);
#pragma unroll
    for (int r = 0; r < 8; ++r) p[r] = (__bf16)acc1[r];
    *(bf16x8*)(At + (size_t)e * DH_ + d0) = p;
  }
}

// ---------------------------------------------------------------------------
// K5: out = q_s[4096x64] @ attn[64x64] per (b,h); y bf16 [b][n][512].
// grid (32, 64), block 256; wave = 16 rows x 64 cols, K=64.
// LDS-staged transpose for coalesced row-major stores.
// ---------------------------------------------------------------------------
__global__ void k5_out(const __bf16* __restrict__ q_s, const __bf16* __restrict__ attnT,
                       __bf16* __restrict__ y) {
  const int b = blockIdx.y >> 3, h = blockIdx.y & 7;
  const int lane = threadIdx.x & 31, wave = threadIdx.x >> 5;
  const int row0 = blockIdx.x * 128 + wave * 16;
  const __bf16* A0 = q_s + ((size_t)b * N_ + row0) * HID_ + h * DH_;
  const __bf16* B0 = attnT + (size_t)(b * HEADS_ + h) * DH_ * DH_;
  f32x8 acc[4] = {};
  for (int k = 0; k < DH_; k += 32) {
    bf16x16 a = load_a_frag(A0 + k, HID_, lane);
#pragma unroll
    for (int j = 0; j < 4; ++j) {
      bf16x16 bb = load_b_frag(B0 + (size_t)(j * 16) * DH_ + k, DH_, lane);
      acc[j] = wmma_bf16(a, bb, acc[j]);
    }
  }
  __shared__ __bf16 st[8][16][64];
  const int rr = ((lane >> 4) << 3);
#pragma unroll
  for (int j = 0; j < 4; ++j)
#pragma unroll
    for (int r = 0; r < 8; ++r)
      st[wave][rr + r][j * 16 + (lane & 15)] = (__bf16)acc[j][r];
  __syncthreads();
  __bf16* Y0 = y + ((size_t)b * N_ + row0) * HID_ + h * DH_;
  for (int r = 0; r < 16; ++r) {
    unsigned int v = ((const unsigned int*)&st[wave][r][0])[lane];
    *(unsigned int*)(Y0 + (size_t)r * HID_ + lane * 2) = v;
  }
}

// ---------------------------------------------------------------------------
// K6: conv (GEMM [4096x512]x[512x256]) + bias + rmsnorm over 256 + g_ffn + xn residual.
// grid (256, 8), block 256; workgroup tile 16 rows x 256 cols (full channel span),
// wave = 2 col-tiles. Fused row-wise sumsq via shfl + cross-wave LDS reduction.
// Output fp32 [b][256][4096] via LDS transpose + float4 stores.
// ---------------------------------------------------------------------------
__global__ void k6_conv_norm(const __bf16* __restrict__ y, const __bf16* __restrict__ Wc,
                             const float* __restrict__ conv_b, const float* __restrict__ g_ffn,
                             const float* __restrict__ xn, float* __restrict__ out) {
  const int b = blockIdx.y;
  const int lane = threadIdx.x & 31, wave = threadIdx.x >> 5;
  const int row0 = blockIdx.x * 16;
  const int c0 = wave * 32;
  const __bf16* A0 = y + ((size_t)b * N_ + row0) * HID_;

  f32x8 acc[2] = {};
  for (int k = 0; k < HID_; k += 32) {
    bf16x16 a = load_a_frag(A0 + k, HID_, lane);
    acc[0] = wmma_bf16(a, load_b_frag(Wc + (size_t)c0 * HID_ + k, HID_, lane), acc[0]);
    acc[1] = wmma_bf16(a, load_b_frag(Wc + (size_t)(c0 + 16) * HID_ + k, HID_, lane), acc[1]);
  }
#pragma unroll
  for (int j = 0; j < 2; ++j) {
    float bias = conv_b[c0 + j * 16 + (lane & 15)];
#pragma unroll
    for (int r = 0; r < 8; ++r) acc[j][r] += bias;
  }
  // per-row sum of squares: reduce 32 cols within wave, then across 8 waves
  float ss[8];
#pragma unroll
  for (int r = 0; r < 8; ++r) ss[r] = acc[0][r] * acc[0][r] + acc[1][r] * acc[1][r];
  for (int m = 1; m < 16; m <<= 1)
#pragma unroll
    for (int r = 0; r < 8; ++r) ss[r] += __shfl_xor(ss[r], m, 32);

  __shared__ float lss[8][16];
  __shared__ float rowinv[16];
  if ((lane & 15) == 0) {
    int rb = ((lane >> 4) << 3);
#pragma unroll
    for (int r = 0; r < 8; ++r) lss[wave][rb + r] = ss[r];
  }
  __syncthreads();
  if (threadIdx.x < 16) {
    float s = 0.f;
#pragma unroll
    for (int w = 0; w < 8; ++w) s += lss[w][threadIdx.x];
    rowinv[threadIdx.x] = 16.0f / fmaxf(sqrtf(s), 1e-12f);   // sqrt(256)=16
  }
  __syncthreads();

  __shared__ float st[256][20];   // padded: 20 % 4 == 0 (f32x4 aligned), conflict-free
  const int rb = ((lane >> 4) << 3);
#pragma unroll
  for (int j = 0; j < 2; ++j) {
    int o = c0 + j * 16 + (lane & 15);
    float g = g_ffn[o];
#pragma unroll
    for (int r = 0; r < 8; ++r) {
      int row = rb + r;
      st[o][row] = acc[j][r] * rowinv[row] * g;
    }
  }
  __syncthreads();

  const float* xnp = xn + ((size_t)b * QC_ + threadIdx.x) * N_ + row0;
  float* op = out + ((size_t)b * QC_ + threadIdx.x) * N_ + row0;
#pragma unroll
  for (int i = 0; i < 16; i += 4) {
    f32x4 xv = *(const f32x4*)(xnp + i);
    f32x4 o;
#pragma unroll
    for (int u = 0; u < 4; ++u) o[u] = st[threadIdx.x][i + u] + xv[u];
    *(f32x4*)(op + i) = o;
  }
}

// ---------------------------------------------------------------------------
// Host launcher
// ---------------------------------------------------------------------------
extern "C" void kernel_launch(void* const* d_in, const int* in_sizes, int n_in,
                              void* d_out, int out_size, void* d_ws, size_t ws_size,
                              hipStream_t stream) {
  const float* x     = (const float*)d_in[0];
  const float* Wq    = (const float*)d_in[1];
  const float* Wk    = (const float*)d_in[2];
  const float* Wv    = (const float*)d_in[3];
  const float* convw = (const float*)d_in[4];
  const float* convb = (const float*)d_in[5];
  const float* gffn  = (const float*)d_in[6];
  const float* gnorm = (const float*)d_in[7];

  // workspace layout (bytes)
  constexpr size_t SZ_T   = (size_t)B_ * N_ * QC_ * 2;     // t bf16       16.8 MB
  constexpr size_t SZ_XN  = (size_t)B_ * QC_ * N_ * 4;     // xn fp32      33.5 MB
  constexpr size_t SZ_QS  = (size_t)B_ * N_ * HID_ * 2;    // q_s bf16     33.5 MB
  constexpr size_t SZ_KT  = (size_t)B_ * HID_ * N_ * 4;    // kT fp32      67  MB
  constexpr size_t SZ_VT  = (size_t)B_ * HID_ * N_ * 2;    // vT bf16      33.5 MB
  constexpr size_t SZ_ET  = (size_t)B_ * HID_ * N_ * 2;    // Et bf16      33.5 MB
  constexpr size_t SZ_AT  = (size_t)B_ * HEADS_ * DH_ * DH_ * 2; // attnT  0.5 MB
  constexpr size_t SZ_W   = (size_t)HID_ * QC_ * 2;        // one bf16 weight

  char* ws = (char*)d_ws;
  size_t off = 0;
  __bf16* t    = (__bf16*)(ws + off); off += SZ_T;
  float*  xn   = (float*) (ws + off); off += SZ_XN;
  __bf16* q_s  = (__bf16*)(ws + off); off += SZ_QS;
  float*  kT   = (float*) (ws + off); off += SZ_KT;
  __bf16* vT   = (__bf16*)(ws + off); off += SZ_VT;
  __bf16* Et   = (__bf16*)(ws + off); off += SZ_ET;
  __bf16* attnT= (__bf16*)(ws + off); off += SZ_AT;
  __bf16* Wqb  = (__bf16*)(ws + off); off += SZ_W;
  __bf16* Wkb  = (__bf16*)(ws + off); off += SZ_W;
  __bf16* Wvb  = (__bf16*)(ws + off); off += SZ_W;
  __bf16* Wcb  = (__bf16*)(ws + off); off += SZ_W;
  __bf16* y    = (__bf16*)kT;   // kT is dead after K3; reuse for y bf16 (33.5 MB < 67 MB)

  const int wn = HID_ * QC_;    // 131072 elements each
  k0_cvt_bf16<<<(wn + 255) / 256, 256, 0, stream>>>(Wq,    Wqb, wn);
  k0_cvt_bf16<<<(wn + 255) / 256, 256, 0, stream>>>(Wk,    Wkb, wn);
  k0_cvt_bf16<<<(wn + 255) / 256, 256, 0, stream>>>(Wv,    Wvb, wn);
  k0_cvt_bf16<<<(wn + 255) / 256, 256, 0, stream>>>(convw, Wcb, wn);

  k1_rmsnorm_t<<<dim3(N_ / 256, B_), 256, 0, stream>>>(x, gnorm, xn, t);
  k2_qkv<<<dim3(N_ / 128, HID_ / 64, B_ * 3), 256, 0, stream>>>(t, Wqb, Wkb, Wvb,
                                                                q_s, kT, vT);
  k3_ksoftmax<<<B_ * HID_, 256, 0, stream>>>(kT, Et);
  k4_attn<<<B_ * HEADS_, 256, 0, stream>>>(Et, vT, attnT);
  k5_out<<<dim3(N_ / 128, B_ * HEADS_), 256, 0, stream>>>(q_s, attnT, y);
  k6_conv_norm<<<dim3(N_ / 16, B_), 256, 0, stream>>>(y, Wcb, convb, gffn, xn,
                                                      (float*)d_out);
}
